// GNN_1254130451159
// MI455X (gfx1250) — compile-verified, matrix operand order checked by reading, not run on hardware
//
#include <hip/hip_runtime.h>
#include <hip/hip_bf16.h>

// ---------------- problem constants (from reference) ----------------
#define BB    512
#define NP    200
#define EPER  2000
#define EDG   (BB*EPER)        // 1,024,000
#define EMB   9
#define HID   128
#define K1    160              // ceil(0.8*200)
#define K2    128              // ceil(0.8*160)
#define K3    103              // ceil(0.8*128)
#define N0    (BB*NP)          // 102400
#define N1    (BB*K1)          // 81920
#define N2    (BB*K2)          // 65536
#define N3    (BB*K3)          // 52736

// ---------------- types ----------------
typedef __attribute__((ext_vector_type(16))) __bf16 v16bf;
typedef __attribute__((ext_vector_type(8)))  __bf16 v8bf;
typedef __attribute__((ext_vector_type(8)))  float  v8f;

__device__ __forceinline__ __bf16 f2bf(float f) {
    unsigned u = __builtin_bit_cast(unsigned, f);
    unsigned r = u + 0x7FFFu + ((u >> 16) & 1u);   // round-to-nearest-even
    unsigned short h = (unsigned short)(r >> 16);
    return __builtin_bit_cast(__bf16, h);
}

// ---------------- small utility kernels ----------------
__global__ void k_f32_to_bf16(const float* __restrict__ in, __bf16* __restrict__ out, long n) {
    long i = (long)blockIdx.x * blockDim.x + threadIdx.x;
    if (i < n) out[i] = f2bf(in[i]);
}

__global__ void k_init_edges(const int* __restrict__ ei, int* __restrict__ src,
                             int* __restrict__ dst, float* __restrict__ valid) {
    int e = blockIdx.x * blockDim.x + threadIdx.x;
    if (e >= EDG) return;
    src[e] = ei[e];
    dst[e] = ei[EDG + e];
    valid[e] = 1.0f;
}

__global__ void k_gather_emb(const int* __restrict__ node_ids, const float* __restrict__ emb,
                             float* __restrict__ x0) {
    long i = (long)blockIdx.x * blockDim.x + threadIdx.x;   // over N0*EMB
    if (i >= (long)N0 * EMB) return;
    long n = i / EMB, f = i % EMB;
    x0[i] = emb[(long)node_ids[n] * EMB + f];
}

// ---------------- conv1 (EMB=9, plain FMA) ----------------
__global__ void k_scatter9(const float* __restrict__ x0, const int* __restrict__ src,
                           const int* __restrict__ dst, float* __restrict__ acc,
                           float* __restrict__ cnt) {
    int e = blockIdx.x * blockDim.x + threadIdx.x;
    if (e >= EDG) return;
    int s = src[e], d = dst[e];
    #pragma unroll
    for (int f = 0; f < EMB; ++f)
        atomicAdd(&acc[(long)d * EMB + f], x0[(long)s * EMB + f]);
    atomicAdd(&cnt[d], 1.0f);
}

__global__ void k_conv1(const float* __restrict__ x0, const float* __restrict__ acc,
                        const float* __restrict__ cnt, const float* __restrict__ w1n,
                        const float* __restrict__ w1r, const float* __restrict__ b1,
                        float* __restrict__ out) {
    long i = (long)blockIdx.x * blockDim.x + threadIdx.x;   // over N0*HID
    if (i >= (long)N0 * HID) return;
    int node = (int)(i >> 7), h = (int)(i & 127);
    float c = fmaxf(cnt[node], 1.0f);
    float s = b1[h];
    #pragma unroll
    for (int f = 0; f < EMB; ++f)
        s += (acc[(long)node * EMB + f] / c) * w1n[h * EMB + f]
           + x0[(long)node * EMB + f] * w1r[h * EMB + f];
    out[i] = fmaxf(s, 0.0f);
}

// ---------------- pooling ----------------
__global__ void k_pnorm(const float* __restrict__ p, float* __restrict__ out) {
    __shared__ float sm[HID];
    int t = threadIdx.x;
    float v = p[t];
    sm[t] = v * v;
    __syncthreads();
    for (int s = HID / 2; s > 0; s >>= 1) {
        if (t < s) sm[t] += sm[t + s];
        __syncthreads();
    }
    if (t == 0) out[0] = sqrtf(sm[0]);
}

__global__ void k_score(const float* __restrict__ x, const float* __restrict__ p,
                        const float* __restrict__ pn, float* __restrict__ score, int n) {
    int i = blockIdx.x * blockDim.x + threadIdx.x;
    if (i >= n) return;
    float s = 0.f;
    #pragma unroll 4
    for (int f = 0; f < HID; ++f) s += x[(long)i * HID + f] * p[f];
    score[i] = tanhf(s / pn[0]);
}

// one block per graph; bitonic sort 256 (score,idx) pairs descending, keep top k
__global__ void k_topk_pool(const float* __restrict__ score, const float* __restrict__ xin,
                            float* __restrict__ xout, int* __restrict__ mapping,
                            int np_cur, int k) {
    __shared__ float sv[256];
    __shared__ int   si[256];
    int b = blockIdx.x, t = threadIdx.x;
    sv[t] = (t < np_cur) ? score[b * np_cur + t] : -1e30f;
    si[t] = t;
    __syncthreads();
    for (int size = 2; size <= 256; size <<= 1) {
        for (int stride = size >> 1; stride > 0; stride >>= 1) {
            int p = t ^ stride;
            if (p > t) {
                bool desc = ((t & size) == 0);
                float a = sv[t], c = sv[p];
                bool swap = desc ? (a < c) : (a > c);
                if (swap) {
                    sv[t] = c; sv[p] = a;
                    int tmp = si[t]; si[t] = si[p]; si[p] = tmp;
                }
            }
            __syncthreads();
        }
    }
    if (t < k) {
        int oldid = b * np_cur + si[t];
        int newid = b * k + t;
        float v = sv[t];
        mapping[oldid] = newid;
        #pragma unroll 4
        for (int f = 0; f < HID; ++f)
            xout[(long)newid * HID + f] = xin[(long)oldid * HID + f] * v;
    }
}

__global__ void k_remap(const int* __restrict__ mapping, int* __restrict__ src,
                        int* __restrict__ dst, float* __restrict__ valid) {
    int e = blockIdx.x * blockDim.x + threadIdx.x;
    if (e >= EDG) return;
    int ns = mapping[src[e]], nd = mapping[dst[e]];
    float v = valid[e];
    if (ns < 0 || nd < 0) v = 0.0f;
    valid[e] = v;
    src[e] = ns < 0 ? 0 : ns;
    dst[e] = nd < 0 ? 0 : nd;
}

// global max+mean pool, accumulated into h (x1+x2+x3)
__global__ void k_pool(const float* __restrict__ x, float* __restrict__ hpool, int k) {
    int b = blockIdx.x, f = threadIdx.x;   // 128 threads
    float mx = -1e30f, sm = 0.f;
    for (int j = 0; j < k; ++j) {
        float v = x[(long)(b * k + j) * HID + f];
        mx = fmaxf(mx, v);
        sm += v;
    }
    hpool[b * 2 * HID + f]       += mx;
    hpool[b * 2 * HID + HID + f] += sm / (float)k;
}

// ---------------- conv2/3 aggregation ----------------
__global__ void k_scatter128(const float* __restrict__ x, const int* __restrict__ src,
                             const int* __restrict__ dst, const float* __restrict__ valid,
                             float* __restrict__ acc, float* __restrict__ cnt) {
    int e = blockIdx.x, f = threadIdx.x;   // 128 threads per edge
    float v = valid[e];
    if (v <= 0.0f) return;
    int s = src[e], d = dst[e];
    atomicAdd(&acc[(long)d * HID + f], x[(long)s * HID + f]);
    if (f == 0) atomicAdd(&cnt[d], v);
}

__global__ void k_prep_bf16(const float* __restrict__ x, const float* __restrict__ acc,
                            const float* __restrict__ cnt, __bf16* __restrict__ xbf,
                            __bf16* __restrict__ meanbf, int n) {
    long i = (long)blockIdx.x * blockDim.x + threadIdx.x;   // over n*HID
    if (i >= (long)n * HID) return;
    int node = (int)(i >> 7);
    float c = fmaxf(cnt[node], 1.0f);
    meanbf[i] = f2bf(acc[i] / c);
    xbf[i]    = f2bf(x[i]);
}

// ---------------- WMMA GEMM: out = relu( A1@W1^T [+ A2@W2^T] + bias ) ----------------
// CDNA5 16-bit A/B tile striping (ISA 7.12.2): lane L holds row (L&15); per lane
// 16 bf16 values: K = kbase + {0..7} and kbase + 16 + {0..7}, kbase = (L>=16)?8:0.
__device__ __forceinline__ v16bf load_tile16(const __bf16* __restrict__ tile, int ld,
                                             int lane, int kbase) {
    int row = lane & 15;
    int kb = kbase + ((lane >> 4) << 3);
    const __bf16* p = tile + (long)row * ld + kb;
    v8bf lo = *(const v8bf*)(p);         // K = kb .. kb+7
    v8bf hi = *(const v8bf*)(p + 16);    // K = kb+16 .. kb+23
    return __builtin_shufflevector(lo, hi, 0,1,2,3,4,5,6,7,8,9,10,11,12,13,14,15);
}

// DUAL / KDIM / NDIM are compile-time -> fully unrolled K loop, no uniform branches,
// back-to-back v_wmma_f32_16x16x32_bf16 with clause-packed b128 loads.
template <bool DUAL, int KDIM, int NDIM>
__global__ void k_gemm(const __bf16* __restrict__ A1, const __bf16* __restrict__ W1,
                       const __bf16* __restrict__ A2, const __bf16* __restrict__ W2,
                       const float* __restrict__ bias, float* __restrict__ out) {
    // one wave (32 threads) per 16x16 output tile
    constexpr int NTN = NDIM >> 4;
    int tile = blockIdx.x;
    int tm = tile / NTN, tn = tile - tm * NTN;
    int lane = threadIdx.x;
    const __bf16* a1t = A1 + (long)tm * 16 * KDIM;
    const __bf16* w1t = W1 + (long)tn * 16 * KDIM;   // W row-major (N x K): B[k][n] = W[n][k]
    const __bf16* a2t = DUAL ? (A2 + (long)tm * 16 * KDIM) : nullptr;
    const __bf16* w2t = DUAL ? (W2 + (long)tn * 16 * KDIM) : nullptr;

    v8f c = {};
    constexpr int NKS = KDIM >> 5;
    #pragma unroll
    for (int ks = 0; ks < NKS; ++ks) {
        const int kb = ks << 5;
        if (ks + 1 < NKS) {  // prefetch next K-slab of A1 -> global_prefetch_b8
            __builtin_prefetch(a1t + (long)(lane & 15) * KDIM + kb + 32, 0, 3);
        }
        v16bf a = load_tile16(a1t, KDIM, lane, kb);
        v16bf b = load_tile16(w1t, KDIM, lane, kb);
        c = __builtin_amdgcn_wmma_f32_16x16x32_bf16(false, a, false, b,
                                                    (short)0, c, false, false);
        if (DUAL) {
            v16bf a2 = load_tile16(a2t, KDIM, lane, kb);
            v16bf b2 = load_tile16(w2t, KDIM, lane, kb);
            c = __builtin_amdgcn_wmma_f32_16x16x32_bf16(false, a2, false, b2,
                                                        (short)0, c, false, false);
        }
    }
    // C/D layout: VGPR r -> (M = tm*16 + r + (lane>=16 ? 8 : 0), N = tn*16 + (lane&15))
    int n = (tn << 4) + (lane & 15);
    float bv = bias[n];
    int mbase = (tm << 4) + ((lane >> 4) << 3);
    #pragma unroll
    for (int r = 0; r < 8; ++r) {
        float v = c[r] + bv;
        out[(long)(mbase + r) * NDIM + n] = fmaxf(v, 0.0f);
    }
}

// ---------------- final 64->1 + sigmoid ----------------
__global__ void k_final(const float* __restrict__ m2, const float* __restrict__ lw3,
                        const float* __restrict__ lb3, float* __restrict__ out) {
    int b = blockIdx.x * blockDim.x + threadIdx.x;
    if (b >= BB) return;
    float s = lb3[0];
    #pragma unroll 4
    for (int i = 0; i < 64; ++i) s += m2[b * 64 + i] * lw3[i];
    out[b] = 1.0f / (1.0f + expf(-s));
}

// ---------------- host side ----------------
static inline char* carve(char*& p, size_t bytes) {
    char* r = p;
    p += (bytes + 255) & ~(size_t)255;
    return r;
}

extern "C" void kernel_launch(void* const* d_in, const int* in_sizes, int n_in,
                              void* d_out, int out_size, void* d_ws, size_t ws_size,
                              hipStream_t stream) {
    const int*   node_ids = (const int*)d_in[0];
    const int*   edge_idx = (const int*)d_in[1];
    // d_in[2] = batch (unused; ids are contiguous per graph)
    const float* emb = (const float*)d_in[3];
    const float* w1n = (const float*)d_in[4];
    const float* w1r = (const float*)d_in[5];
    const float* b1  = (const float*)d_in[6];
    const float* w2n = (const float*)d_in[7];
    const float* w2r = (const float*)d_in[8];
    const float* b2  = (const float*)d_in[9];
    const float* w3n = (const float*)d_in[10];
    const float* w3r = (const float*)d_in[11];
    const float* b3  = (const float*)d_in[12];
    const float* p1  = (const float*)d_in[13];
    const float* p2  = (const float*)d_in[14];
    const float* p3  = (const float*)d_in[15];
    const float* lw1 = (const float*)d_in[16];
    const float* lb1 = (const float*)d_in[17];
    const float* lw2 = (const float*)d_in[18];
    const float* lb2 = (const float*)d_in[19];
    const float* lw3 = (const float*)d_in[20];
    const float* lb3 = (const float*)d_in[21];
    float* out = (float*)d_out;

    char* ws = (char*)d_ws;
    float*  bufA    = (float*) carve(ws, (size_t)N0 * HID * 4);
    float*  bufB    = (float*) carve(ws, (size_t)N0 * HID * 4);
    float*  acc128  = (float*) carve(ws, (size_t)N1 * HID * 4);
    __bf16* xbf     = (__bf16*)carve(ws, (size_t)N1 * HID * 2);
    __bf16* meanbf  = (__bf16*)carve(ws, (size_t)N1 * HID * 2);
    float*  x0      = (float*) carve(ws, (size_t)N0 * EMB * 4);
    float*  acc9    = (float*) carve(ws, (size_t)N0 * EMB * 4);
    float*  cnt     = (float*) carve(ws, (size_t)N0 * 4);
    float*  score   = (float*) carve(ws, (size_t)N0 * 4);
    int*    mapping = (int*)   carve(ws, (size_t)N0 * 4);
    int*    srcc    = (int*)   carve(ws, (size_t)EDG * 4);
    int*    dstc    = (int*)   carve(ws, (size_t)EDG * 4);
    float*  valid   = (float*) carve(ws, (size_t)EDG * 4);
    float*  hpool   = (float*) carve(ws, (size_t)BB * 2 * HID * 4);
    float*  pn      = (float*) carve(ws, 256);
    __bf16* w2nbf   = (__bf16*)carve(ws, (size_t)HID * HID * 2);
    __bf16* w2rbf   = (__bf16*)carve(ws, (size_t)HID * HID * 2);
    __bf16* w3nbf   = (__bf16*)carve(ws, (size_t)HID * HID * 2);
    __bf16* w3rbf   = (__bf16*)carve(ws, (size_t)HID * HID * 2);
    __bf16* lw1bf   = (__bf16*)carve(ws, (size_t)HID * 2 * HID * 2);
    __bf16* lw2bf   = (__bf16*)carve(ws, (size_t)64 * HID * 2);
    __bf16* hbf     = (__bf16*)carve(ws, (size_t)BB * 2 * HID * 2);
    float*  m1      = (float*) carve(ws, (size_t)BB * HID * 4);
    __bf16* m1bf    = (__bf16*)carve(ws, (size_t)BB * HID * 2);
    float*  m2      = (float*) carve(ws, (size_t)BB * 64 * 4);

    const int T = 256;
    #define GRID1(n) dim3((unsigned)(((long)(n) + T - 1) / T)), dim3(T)

    // weight conversions (deterministic each call)
    k_f32_to_bf16<<<GRID1(HID*HID), 0, stream>>>(w2n, w2nbf, (long)HID*HID);
    k_f32_to_bf16<<<GRID1(HID*HID), 0, stream>>>(w2r, w2rbf, (long)HID*HID);
    k_f32_to_bf16<<<GRID1(HID*HID), 0, stream>>>(w3n, w3nbf, (long)HID*HID);
    k_f32_to_bf16<<<GRID1(HID*HID), 0, stream>>>(w3r, w3rbf, (long)HID*HID);
    k_f32_to_bf16<<<GRID1(HID*2*HID), 0, stream>>>(lw1, lw1bf, (long)HID*2*HID);
    k_f32_to_bf16<<<GRID1(64*HID), 0, stream>>>(lw2, lw2bf, (long)64*HID);

    hipMemsetAsync(hpool, 0, (size_t)BB * 2 * HID * 4, stream);

    // ---- stage 0: embed + conv1 ----
    k_init_edges<<<GRID1(EDG), 0, stream>>>(edge_idx, srcc, dstc, valid);
    k_gather_emb<<<GRID1((long)N0*EMB), 0, stream>>>(node_ids, emb, x0);
    hipMemsetAsync(acc9, 0, (size_t)N0 * EMB * 4, stream);
    hipMemsetAsync(cnt,  0, (size_t)N0 * 4, stream);
    k_scatter9<<<GRID1(EDG), 0, stream>>>(x0, srcc, dstc, acc9, cnt);
    k_conv1<<<GRID1((long)N0*HID), 0, stream>>>(x0, acc9, cnt, w1n, w1r, b1, bufA);

    // ---- pool 1 ----
    k_pnorm<<<dim3(1), dim3(HID), 0, stream>>>(p1, pn);
    k_score<<<GRID1(N0), 0, stream>>>(bufA, p1, pn, score, N0);
    hipMemsetAsync(mapping, 0xFF, (size_t)N0 * 4, stream);
    k_topk_pool<<<dim3(BB), dim3(256), 0, stream>>>(score, bufA, bufB, mapping, NP, K1);
    k_remap<<<GRID1(EDG), 0, stream>>>(mapping, srcc, dstc, valid);
    k_pool<<<dim3(BB), dim3(HID), 0, stream>>>(bufB, hpool, K1);

    // ---- conv2 (WMMA) ----
    hipMemsetAsync(acc128, 0, (size_t)N1 * HID * 4, stream);
    hipMemsetAsync(cnt,    0, (size_t)N1 * 4, stream);
    k_scatter128<<<dim3(EDG), dim3(HID), 0, stream>>>(bufB, srcc, dstc, valid, acc128, cnt);
    k_prep_bf16<<<GRID1((long)N1*HID), 0, stream>>>(bufB, acc128, cnt, xbf, meanbf, N1);
    k_gemm<true, HID, HID><<<dim3((N1/16)*(HID/16)), dim3(32), 0, stream>>>(
        meanbf, w2nbf, xbf, w2rbf, b2, bufA);

    // ---- pool 2 ----
    k_pnorm<<<dim3(1), dim3(HID), 0, stream>>>(p2, pn);
    k_score<<<GRID1(N1), 0, stream>>>(bufA, p2, pn, score, N1);
    hipMemsetAsync(mapping, 0xFF, (size_t)N1 * 4, stream);
    k_topk_pool<<<dim3(BB), dim3(256), 0, stream>>>(score, bufA, bufB, mapping, K1, K2);
    k_remap<<<GRID1(EDG), 0, stream>>>(mapping, srcc, dstc, valid);
    k_pool<<<dim3(BB), dim3(HID), 0, stream>>>(bufB, hpool, K2);

    // ---- conv3 (WMMA) ----
    hipMemsetAsync(acc128, 0, (size_t)N2 * HID * 4, stream);
    hipMemsetAsync(cnt,    0, (size_t)N2 * 4, stream);
    k_scatter128<<<dim3(EDG), dim3(HID), 0, stream>>>(bufB, srcc, dstc, valid, acc128, cnt);
    k_prep_bf16<<<GRID1((long)N2*HID), 0, stream>>>(bufB, acc128, cnt, xbf, meanbf, N2);
    k_gemm<true, HID, HID><<<dim3((N2/16)*(HID/16)), dim3(32), 0, stream>>>(
        meanbf, w3nbf, xbf, w3rbf, b3, bufA);

    // ---- pool 3 ----
    k_pnorm<<<dim3(1), dim3(HID), 0, stream>>>(p3, pn);
    k_score<<<GRID1(N2), 0, stream>>>(bufA, p3, pn, score, N2);
    hipMemsetAsync(mapping, 0xFF, (size_t)N2 * 4, stream);
    k_topk_pool<<<dim3(BB), dim3(256), 0, stream>>>(score, bufA, bufB, mapping, K2, K3);
    k_pool<<<dim3(BB), dim3(HID), 0, stream>>>(bufB, hpool, K3);

    // ---- readout MLP (WMMA for layers 1-2) ----
    k_f32_to_bf16<<<GRID1((long)BB*2*HID), 0, stream>>>(hpool, hbf, (long)BB*2*HID);
    k_gemm<false, 2*HID, HID><<<dim3((BB/16)*(HID/16)), dim3(32), 0, stream>>>(
        hbf, lw1bf, nullptr, nullptr, lb1, m1);
    k_f32_to_bf16<<<GRID1((long)BB*HID), 0, stream>>>(m1, m1bf, (long)BB*HID);
    k_gemm<false, HID, 64><<<dim3((BB/16)*(64/16)), dim3(32), 0, stream>>>(
        m1bf, lw2bf, nullptr, nullptr, lb2, m2);
    k_final<<<GRID1(BB), 0, stream>>>(m2, lw3, lb3, out);

    #undef GRID1
    (void)in_sizes; (void)n_in; (void)out_size; (void)ws_size;
}
